// GraphConvolution_36928128811026
// MI455X (gfx1250) — matrix operand order repaired and checked by reading, not compiled, not run on hardware
//
#include <hip/hip_runtime.h>

// Problem constants (fixed by the reference).
#define NU_   16384
#define NV_   8192
#define UF_   1024
#define HID_  512
#define NCLS_ 5

typedef __attribute__((ext_vector_type(16))) __bf16 v16bf;
typedef __attribute__((ext_vector_type(2)))  __bf16 v2bf;
typedef __attribute__((ext_vector_type(8)))  float  v8f;
typedef __attribute__((ext_vector_type(2)))  float  v2f;
typedef __attribute__((ext_vector_type(4)))  unsigned int v4u;
typedef __attribute__((ext_vector_type(8)))  unsigned int v8u;

union Frag16 {
  v16bf v;
  unsigned short s[16];
  unsigned int   u[8];
};

// f32 -> bf16 (RNE) via native __bf16 (lowers to v_cvt_pk_bf16_f32).
__device__ __forceinline__ unsigned short bf16_bits(float f) {
  union { __bf16 h; unsigned short s; } c;
  c.h = (__bf16)f;
  return c.s;
}

// Two f32 -> one dword of two bf16 in a single packed convert.
__device__ __forceinline__ unsigned int pack_bf16x2(float a, float b) {
  union { v2bf v; unsigned int u; } cv;
  v2f in;
  in.x = a;
  in.y = b;
  cv.v = __builtin_convertvector(in, v2bf);   // v_cvt_pk_bf16_f32
  return cv.u;
}

// --------------------------------------------------------------------------
// Kernel 1: summed class weights -> bf16, stored COLUMN-MAJOR [HID][UF]
// so downstream TDM tiles are k-contiguous.
// --------------------------------------------------------------------------
__global__ void sum_weights_kernel(const float* __restrict__ uW,
                                   const float* __restrict__ vW,
                                   const int* __restrict__ rptr,
                                   unsigned short* __restrict__ uw16,
                                   unsigned short* __restrict__ vw16) {
  const int per = UF_ * HID_;
  int gid = blockIdx.x * blockDim.x + threadIdx.x;
  int r = *rptr;
  if (r > NCLS_ - 1) r = NCLS_ - 1;
  if (r < 0) r = 0;
  const float* src;
  unsigned short* dst;
  int idx;
  if (gid < per) { src = uW; dst = uw16; idx = gid; }
  else           { src = vW; dst = vw16; idx = gid - per; }
  float s = 0.f;
  for (int c = 0; c <= r; ++c) s += src[(size_t)c * per + idx];
  int k = idx / HID_;            // source is [UF][HID] row-major
  int n = idx % HID_;
  dst[(size_t)n * UF_ + k] = bf16_bits(s);   // transposed store
}

// --------------------------------------------------------------------------
// Kernel 2: tiled WMMA bf16 GEMM.
//   A   : f32 [M][lda] (AT: accessed as A[k][m]), staged regs->convert->LDS.
//   B16 : bf16 COLUMN-MAJOR [512][K] (k contiguous); per-tile copy done by
//         the Tensor Data Mover (tensor_load_to_lds) with LDS padding set to
//         reproduce Bs[BN][BK+PAD]. Tracked by TENSORcnt.
//   EPI 0: out16[n][m] (column-major, ldout=M) = bf16(acc + aux[n])
//   EPI 1: outf [m][n] (row-major, pitch N)    = acc * aux[deg_off + m]
// Block = 256 threads = 8 waves (wave32); block tile 128x128, K-tile 32.
// Wave grid 4(M) x 2(N); each wave owns 32x64 = 2x4 WMMA 16x16 tiles.
// All dims are exact multiples of the tiles (fixed problem sizes).
// --------------------------------------------------------------------------
template <bool AT, int EPI>
__global__ __launch_bounds__(256)
void gemm_wmma_kernel(const float* __restrict__ A,
                      const unsigned short* __restrict__ B16,
                      const float* __restrict__ aux,
                      unsigned short* __restrict__ out16,
                      float* __restrict__ outf,
                      int M, int N, int K, int lda, int ldout, int deg_off) {
  constexpr int BM = 128, BN = 128, BK = 32, PAD = 8;
  __shared__ unsigned short As[2][BM][BK + PAD];   // [m][k] bf16
  __shared__ unsigned short Bs[2][BN][BK + PAD];   // [n][k] bf16 (TDM dest)

  const int tid   = threadIdx.x;
  const int lane  = tid & 31;
  const int wid   = tid >> 5;
  const int waveM = wid & 3;
  const int waveN = wid >> 2;
  const int half  = lane >> 4;
  const int ln    = lane & 15;
  const int m0    = blockIdx.y * BM;
  const int n0    = blockIdx.x * BN;

  float4 aR[4];    // register staging: all A loads issue back-to-back

  auto fetchA = [&](int k0) {
    if (!AT) {
#pragma unroll
      for (int i = 0; i < 4; ++i) {
        int idx = tid + i * 256;
        int row = idx >> 3;
        int col = (idx & 7) << 2;
        aR[i] = *reinterpret_cast<const float4*>(
            A + (size_t)(m0 + row) * lda + (k0 + col));
      }
    } else {
      // Each thread: same 4-wide m range at two adjacent k rows (k-pair),
      // so staging can pack (k even, k odd) dwords with v_cvt_pk_bf16_f32.
#pragma unroll
      for (int i = 0; i < 2; ++i) {
        int idx = tid + i * 256;       // 512 = 16 k-pairs x 32 m-groups
        int kk2 = idx >> 5;            // k-pair 0..15
        int mv  = (idx & 31) << 2;     // 4 consecutive m
        aR[2 * i + 0] = *reinterpret_cast<const float4*>(
            A + (size_t)(k0 + 2 * kk2 + 0) * lda + (m0 + mv));
        aR[2 * i + 1] = *reinterpret_cast<const float4*>(
            A + (size_t)(k0 + 2 * kk2 + 1) * lda + (m0 + mv));
      }
    }
  };

  auto stageA = [&](int buf) {
    if (!AT) {
#pragma unroll
      for (int i = 0; i < 4; ++i) {
        int idx = tid + i * 256;
        int row = idx >> 3;
        int col = (idx & 7) << 2;
        uint2 p;
        p.x = pack_bf16x2(aR[i].x, aR[i].y);
        p.y = pack_bf16x2(aR[i].z, aR[i].w);
        *reinterpret_cast<uint2*>(&As[buf][row][col]) = p;
      }
    } else {
#pragma unroll
      for (int i = 0; i < 2; ++i) {
        int idx = tid + i * 256;
        int kk2 = idx >> 5;
        int mv  = (idx & 31) << 2;
        float4 lo = aR[2 * i + 0];     // k = 2*kk2
        float4 hi = aR[2 * i + 1];     // k = 2*kk2 + 1
        *reinterpret_cast<unsigned int*>(&As[buf][mv + 0][2 * kk2]) =
            pack_bf16x2(lo.x, hi.x);
        *reinterpret_cast<unsigned int*>(&As[buf][mv + 1][2 * kk2]) =
            pack_bf16x2(lo.y, hi.y);
        *reinterpret_cast<unsigned int*>(&As[buf][mv + 2][2 * kk2]) =
            pack_bf16x2(lo.z, hi.z);
        *reinterpret_cast<unsigned int*>(&As[buf][mv + 3][2 * kk2]) =
            pack_bf16x2(lo.w, hi.w);
      }
    }
  };

  // Issue one TDM descriptor for the B tile: 2D tile 32(k) x 128(n) of the
  // column-major [512][K] tensor, padded into Bs[buf][128][32+8].
  auto tdmB = [&](int k0, int buf) {
    if (wid == 0) {
      unsigned long long ga =
          (unsigned long long)(size_t)(B16 + (size_t)n0 * K + k0);
      unsigned int lds = (unsigned int)(size_t)(&Bs[buf][0][0]);
      v4u g0;
      g0[0] = 1u;                                          // count=1
      g0[1] = lds;                                         // lds_addr
      g0[2] = (unsigned int)(ga & 0xffffffffu);            // global_addr lo
      g0[3] = (unsigned int)((ga >> 32) & 0x1ffffffu)      // global_addr hi
              | (2u << 30);                                // type=2 (image)
      v8u g1;
      // data_size=2B(code 1), pad_enable, pad_interval: 16 dwords (code 3),
      // pad_amount: 4 dwords (code 3)  -> row 64B + 16B pad = 80B pitch.
      g1[0] = (1u << 16) | (1u << 20) | (3u << 22) | (3u << 25);
      g1[1] = ((unsigned int)K & 0xffffu) << 16;           // tensor_dim0 lo16
      g1[2] = ((unsigned int)K >> 16) | ((unsigned int)HID_ << 16); // dim0 hi | dim1 lo
      g1[3] = ((unsigned int)BK << 16);                    // dim1 hi=0 | tile_dim0=32
      g1[4] = (unsigned int)BN;                            // tile_dim1=128, tile_dim2=0
      g1[5] = (unsigned int)K;                             // tensor_dim0_stride lo32
      g1[6] = 0u;                                          // stride hi | dim1_stride lo
      g1[7] = 0u;
      asm volatile("tensor_load_to_lds %0, %1"
                   :
                   : "s"(g0), "s"(g1)
                   : "memory");
    }
  };

  const v8f vzero = {0.f, 0.f, 0.f, 0.f, 0.f, 0.f, 0.f, 0.f};
  v8f acc[2][4];
#pragma unroll
  for (int i = 0; i < 2; ++i)
#pragma unroll
    for (int j = 0; j < 4; ++j) acc[i][j] = vzero;

  fetchA(0);
  tdmB(0, 0);
  int buf = 0;
  for (int k0 = 0; k0 < K; k0 += BK) {
    stageA(buf);
    if (wid == 0) __builtin_amdgcn_s_wait_tensorcnt((short)0);  // B[buf] landed
    __syncthreads();
    if (k0 + BK < K) {
      fetchA(k0 + BK);           // next A loads issue under the WMMAs
      tdmB(k0 + BK, buf ^ 1);    // next B tile DMA runs under the WMMAs
      if (k0 + 2 * BK < K) {
        const float* pf = AT ? (A + (size_t)(k0 + 2 * BK) * lda + m0)
                             : (A + (size_t)m0 * lda + (k0 + 2 * BK));
        __builtin_prefetch(pf, 0, 1);
      }
    }

    // ---- fragments per ISA 7.12.2 16-bit layouts (aligned dword reads) ----
    Frag16 a[2], b[4];
#pragma unroll
    for (int mt = 0; mt < 2; ++mt) {
      int row = waveM * 32 + mt * 16 + ln;
#pragma unroll
      for (int g = 0; g < 8; ++g) {
        int kb = (g < 4) ? (half * 8 + g * 2) : (16 + half * 8 + (g - 4) * 2);
        a[mt].u[g] = *reinterpret_cast<const unsigned int*>(&As[buf][row][kb]);
      }
    }
#pragma unroll
    for (int nt = 0; nt < 4; ++nt) {
      int col = waveN * 64 + nt * 16 + ln;
#pragma unroll
      for (int g = 0; g < 8; ++g)
        b[nt].u[g] = *reinterpret_cast<const unsigned int*>(
            &Bs[buf][col][(g < 4) ? (half * 8 + g * 2) : (16 + half * 8 + (g - 4) * 2)]);
    }

    // ---- 8x v_wmma_f32_16x16x32_bf16 ----
#pragma unroll
    for (int mt = 0; mt < 2; ++mt)
#pragma unroll
      for (int nt = 0; nt < 4; ++nt)
        acc[mt][nt] = __builtin_amdgcn_wmma_f32_16x16x32_bf16(
            false, a[mt].v, false, b[nt].v, (short)0, acc[mt][nt],
            false, false);
    buf ^= 1;
  }

  // ---- fused epilogue (C/D layout: m = vgpr + 8*half, n = lane&15) ----
#pragma unroll
  for (int mt = 0; mt < 2; ++mt) {
#pragma unroll
    for (int nt = 0; nt < 4; ++nt) {
      int gn = n0 + waveN * 64 + nt * 16 + ln;
      if (EPI == 0) {
        // column-major bf16 out: 8 consecutive m per lane -> one b128 store
        float bia = aux[gn];
        int gm = m0 + waveM * 32 + mt * 16 + 8 * half;
        v4u pk;
        pk[0] = pack_bf16x2(acc[mt][nt][0] + bia, acc[mt][nt][1] + bia);
        pk[1] = pack_bf16x2(acc[mt][nt][2] + bia, acc[mt][nt][3] + bia);
        pk[2] = pack_bf16x2(acc[mt][nt][4] + bia, acc[mt][nt][5] + bia);
        pk[3] = pack_bf16x2(acc[mt][nt][6] + bia, acc[mt][nt][7] + bia);
        *reinterpret_cast<v4u*>(&out16[(size_t)gn * ldout + gm]) = pk;
      } else {
#pragma unroll
        for (int rr = 0; rr < 8; ++rr) {
          int gm = m0 + waveM * 32 + mt * 16 + rr + 8 * half;
          outf[(size_t)gm * N + gn] = acc[mt][nt][rr] * aux[deg_off + gm];
        }
      }
    }
  }
}

// --------------------------------------------------------------------------
extern "C" void kernel_launch(void* const* d_in, const int* in_sizes, int n_in,
                              void* d_out, int out_size, void* d_ws, size_t ws_size,
                              hipStream_t stream) {
  const float* u        = (const float*)d_in[0];
  const float* v        = (const float*)d_in[1];
  const float* adj      = (const float*)d_in[2];
  const float* degree   = (const float*)d_in[3];
  const float* u_weight = (const float*)d_in[4];
  const float* v_weight = (const float*)d_in[5];
  const float* bias     = (const float*)d_in[6];
  const int*   rptr     = (const int*)d_in[7];
  float* out = (float*)d_out;

  // Workspace (bf16, all COLUMN-MAJOR [HID][K], ~27.3 MB total)
  unsigned short* uw16 = (unsigned short*)d_ws;                 // [512][1024]
  unsigned short* vw16 = uw16 + (size_t)UF_ * HID_;             // [512][1024]
  unsigned short* su16 = vw16 + (size_t)UF_ * HID_;             // [512][16384]
  unsigned short* sv16 = su16 + (size_t)NU_ * HID_;             // [512][8192]

  sum_weights_kernel<<<(2 * UF_ * HID_) / 256, 256, 0, stream>>>(
      u_weight, v_weight, rptr, uw16, vw16);

  dim3 blk(256);
  // su^T = bf16(u @ uw + bias)^T : [512][16384]
  gemm_wmma_kernel<false, 0><<<dim3(HID_ / 128, NU_ / 128), blk, 0, stream>>>(
      u, uw16, bias, su16, nullptr, NU_, HID_, UF_, UF_, NU_, 0);
  // sv^T = bf16(v @ vw + bias)^T : [512][8192]
  gemm_wmma_kernel<false, 0><<<dim3(HID_ / 128, NV_ / 128), blk, 0, stream>>>(
      v, vw16, bias, sv16, nullptr, NV_, HID_, UF_, UF_, NV_, 0);
  // out_u = degree[:NU] * (adj @ sv)       -> d_out rows [0, NU)
  gemm_wmma_kernel<false, 1><<<dim3(HID_ / 128, NU_ / 128), blk, 0, stream>>>(
      adj, sv16, degree, nullptr, out, NU_, HID_, NV_, NV_, 0, 0);
  // out_v = degree[NU:] * (adj.T @ su)     -> d_out rows [NU, NU+NV)
  gemm_wmma_kernel<true, 1><<<dim3(HID_ / 128, NV_ / 128), blk, 0, stream>>>(
      adj, su16, degree, nullptr, out + (size_t)NU_ * HID_,
      NV_, HID_, NU_, NV_, 0, NU_);
}
